// P1GCN_40664750359210
// MI455X (gfx1250) — compile-verified, matrix operand order checked by reading, not run on hardware
//
#include <hip/hip_runtime.h>
#include <math.h>

#define N_NODES 50000
#define F_IN 256
#define HID 256
#define N_CLS 40
#define NCLS_PAD 48

typedef __bf16 bf16;
typedef __bf16 v4bf  __attribute__((ext_vector_type(4)));
typedef __bf16 v8bf  __attribute__((ext_vector_type(8)));
typedef __bf16 v16bf __attribute__((ext_vector_type(16)));
typedef float  v8f   __attribute__((ext_vector_type(8)));

// ---------------------------------------------------------------------------
// Edge scatter-add, layer 1: agg[dst] += x[src]   (f32 gather, f32 atomic add)
// 4 edges per 256-thread block; each thread moves a float4 chunk.
// ---------------------------------------------------------------------------
__global__ __launch_bounds__(256)
void scatter_f32_kernel(const float* __restrict__ x,
                        const int* __restrict__ ei, int E,
                        float* __restrict__ agg) {
    int t = threadIdx.x;
    long long e = (long long)blockIdx.x * 4 + (t >> 6);
    if (e >= E) return;
    int f = (t & 63) * 4;
    int s = ei[e];
    int d = ei[(long long)E + e];
    float4 v = *(const float4*)(x + (size_t)s * F_IN + f);
    float* ap = agg + (size_t)d * F_IN + f;
    __hip_atomic_fetch_add(ap + 0, v.x, __ATOMIC_RELAXED, __HIP_MEMORY_SCOPE_AGENT);
    __hip_atomic_fetch_add(ap + 1, v.y, __ATOMIC_RELAXED, __HIP_MEMORY_SCOPE_AGENT);
    __hip_atomic_fetch_add(ap + 2, v.z, __ATOMIC_RELAXED, __HIP_MEMORY_SCOPE_AGENT);
    __hip_atomic_fetch_add(ap + 3, v.w, __ATOMIC_RELAXED, __HIP_MEMORY_SCOPE_AGENT);
}

// ---------------------------------------------------------------------------
// Edge scatter-add, layer 2: agg[dst] += h[src]   (bf16 gather, f32 atomic add)
// ---------------------------------------------------------------------------
__global__ __launch_bounds__(256)
void scatter_bf16_kernel(const bf16* __restrict__ h,
                         const int* __restrict__ ei, int E,
                         float* __restrict__ agg) {
    int t = threadIdx.x;
    long long e = (long long)blockIdx.x * 4 + (t >> 6);
    if (e >= E) return;
    int f = (t & 63) * 4;
    int s = ei[e];
    int d = ei[(long long)E + e];
    v4bf v = *(const v4bf*)(h + (size_t)s * HID + f);
    float* ap = agg + (size_t)d * HID + f;
    __hip_atomic_fetch_add(ap + 0, (float)v[0], __ATOMIC_RELAXED, __HIP_MEMORY_SCOPE_AGENT);
    __hip_atomic_fetch_add(ap + 1, (float)v[1], __ATOMIC_RELAXED, __HIP_MEMORY_SCOPE_AGENT);
    __hip_atomic_fetch_add(ap + 2, (float)v[2], __ATOMIC_RELAXED, __HIP_MEMORY_SCOPE_AGENT);
    __hip_atomic_fetch_add(ap + 3, (float)v[3], __ATOMIC_RELAXED, __HIP_MEMORY_SCOPE_AGENT);
}

// ---------------------------------------------------------------------------
// Streaming f32 -> bf16 convert (4 elements / thread)
// ---------------------------------------------------------------------------
__global__ __launch_bounds__(256)
void cvt_f32_bf16_kernel(const float* __restrict__ in, bf16* __restrict__ out, int n4) {
    int i = blockIdx.x * 256 + threadIdx.x;
    if (i >= n4) return;
    float4 v = ((const float4*)in)[i];
    v4bf o;
    o[0] = (bf16)v.x; o[1] = (bf16)v.y; o[2] = (bf16)v.z; o[3] = (bf16)v.w;
    ((v4bf*)out)[i] = o;
}

// ---------------------------------------------------------------------------
// Weight prepack: W1 [K=256][N=256] f32 row-major -> W1t [N][K] bf16
// ---------------------------------------------------------------------------
__global__ __launch_bounds__(256)
void prep_w1_kernel(const float* __restrict__ W1, bf16* __restrict__ W1t) {
    int n = blockIdx.x;   // 0..255
    int k = threadIdx.x;  // 0..255
    W1t[(size_t)n * 256 + k] = (bf16)W1[(size_t)k * HID + n];
}

// Weight prepack: W2 [K=256][N=40] f32 -> W2t [48][256] bf16 (cols 40..47 = 0)
__global__ __launch_bounds__(256)
void prep_w2_kernel(const float* __restrict__ W2, bf16* __restrict__ W2t) {
    int n = blockIdx.x;   // 0..47
    int k = threadIdx.x;  // 0..255
    float v = (n < N_CLS) ? W2[(size_t)k * N_CLS + n] : 0.0f;
    W2t[(size_t)n * 256 + k] = (bf16)v;
}

// ---------------------------------------------------------------------------
// GEMM1: H = dropout(relu(Agg @ W1 + b1))  -- bf16 WMMA, f32 accumulate.
// Block: 256 threads = 8 waves. Block tile: 16 rows x 128 cols (wave = 16x16).
// A tile staged in LDS, shared across waves. B from global (prepacked W1t).
// ---------------------------------------------------------------------------
__global__ __launch_bounds__(256)
void gemm1_kernel(const bf16* __restrict__ A,    // [N_NODES][256] bf16
                  const bf16* __restrict__ Bt,   // [256][256] bf16 (N-major)
                  const float* __restrict__ b1,
                  const float* __restrict__ dropu,
                  bf16* __restrict__ H) {
    __shared__ bf16 As[16][256];

    int tid = threadIdx.x;
    int i0 = blockIdx.x * 16;

    // Cooperative A stage: 16 rows x 256 cols bf16 (8 KB), 32B per thread.
    {
        int row = tid >> 4;
        int cb  = (tid & 15) * 16;
        const v8bf* src = (const v8bf*)(A + (size_t)(i0 + row) * 256 + cb);
        v8bf* dst = (v8bf*)&As[row][cb];
        dst[0] = src[0];
        dst[1] = src[1];
    }
    __syncthreads();

    int w    = tid >> 5;
    int lane = tid & 31;
    int hsel = lane >> 4;   // K half select (A/B 16-bit fragment layout)
    int m    = lane & 15;
    int j0   = (blockIdx.y * 8 + w) * 16;
    int col  = j0 + m;

    v8f acc;
    float bias = b1[col];
#pragma unroll
    for (int v = 0; v < 8; ++v) acc[v] = bias;

    const bf16* brow = Bt + (size_t)col * 256 + hsel * 8;
    const bf16* arow = &As[m][hsel * 8];

#pragma unroll
    for (int k0 = 0; k0 < 256; k0 += 32) {
        v8bf alo = *(const v8bf*)(arow + k0);
        v8bf ahi = *(const v8bf*)(arow + k0 + 16);
        v8bf blo = *(const v8bf*)(brow + k0);
        v8bf bhi = *(const v8bf*)(brow + k0 + 16);
        v16bf af, bf;
#pragma unroll
        for (int t = 0; t < 8; ++t) {
            af[t] = alo[t]; af[8 + t] = ahi[t];
            bf[t] = blo[t]; bf[8 + t] = bhi[t];
        }
        acc = __builtin_amdgcn_wmma_f32_16x16x32_bf16(
            false, af, false, bf, (short)0, acc, false, false);
    }

    // Epilogue: relu + dropout(0.5 -> x2 scale), store bf16.
    int rbase = i0 + hsel * 8;
#pragma unroll
    for (int v = 0; v < 8; ++v) {
        int row = rbase + v;
        float val = acc[v];
        val = val > 0.0f ? val : 0.0f;
        float u = dropu[(size_t)row * HID + col];
        val = (u > 0.5f) ? val * 2.0f : 0.0f;
        H[(size_t)row * HID + col] = (bf16)val;
    }
}

// ---------------------------------------------------------------------------
// GEMM2: out = log_softmax(Agg2 @ W2 + b2), fused.  128 threads = 4 waves;
// waves 0..2 compute the 16x48 (padded) tile, then block does row softmax.
// ---------------------------------------------------------------------------
__global__ __launch_bounds__(128)
void gemm2_kernel(const bf16* __restrict__ A,    // [N_NODES][256] bf16
                  const bf16* __restrict__ Bt,   // [48][256] bf16
                  const float* __restrict__ b2,
                  float* __restrict__ out) {
    __shared__ bf16 As[16][256];
    __shared__ float Cs[16][NCLS_PAD];

    int tid = threadIdx.x;
    int i0 = blockIdx.x * 16;

    // Cooperative A stage: 128 threads x 32 bf16 (64B) each.
    {
        int row = tid >> 3;
        int cb  = (tid & 7) * 32;
        const v8bf* src = (const v8bf*)(A + (size_t)(i0 + row) * 256 + cb);
        v8bf* dst = (v8bf*)&As[row][cb];
#pragma unroll
        for (int q = 0; q < 4; ++q) dst[q] = src[q];
    }
    __syncthreads();

    int w = tid >> 5;
    if (w < 3) {
        int lane = tid & 31;
        int hsel = lane >> 4;
        int m    = lane & 15;
        int col  = w * 16 + m;     // 0..47

        v8f acc;
        float bias = (col < N_CLS) ? b2[col] : 0.0f;
#pragma unroll
        for (int v = 0; v < 8; ++v) acc[v] = bias;

        const bf16* brow = Bt + (size_t)col * 256 + hsel * 8;
        const bf16* arow = &As[m][hsel * 8];

#pragma unroll
        for (int k0 = 0; k0 < 256; k0 += 32) {
            v8bf alo = *(const v8bf*)(arow + k0);
            v8bf ahi = *(const v8bf*)(arow + k0 + 16);
            v8bf blo = *(const v8bf*)(brow + k0);
            v8bf bhi = *(const v8bf*)(brow + k0 + 16);
            v16bf af, bf;
#pragma unroll
            for (int t = 0; t < 8; ++t) {
                af[t] = alo[t]; af[8 + t] = ahi[t];
                bf[t] = blo[t]; bf[8 + t] = bhi[t];
            }
            acc = __builtin_amdgcn_wmma_f32_16x16x32_bf16(
                false, af, false, bf, (short)0, acc, false, false);
        }

#pragma unroll
        for (int v = 0; v < 8; ++v) {
            Cs[v + 8 * hsel][col] = acc[v];
        }
    }
    __syncthreads();

    // Row-wise log-softmax over the real 40 classes.
    if (tid < 16) {
        int r = tid;
        float mx = -INFINITY;
#pragma unroll 8
        for (int c = 0; c < N_CLS; ++c) mx = fmaxf(mx, Cs[r][c]);
        float s = 0.0f;
#pragma unroll 8
        for (int c = 0; c < N_CLS; ++c) s += __expf(Cs[r][c] - mx);
        float lse = mx + __logf(s);
        float* orow = out + (size_t)(i0 + r) * N_CLS;
#pragma unroll 8
        for (int c = 0; c < N_CLS; ++c) orow[c] = Cs[r][c] - lse;
    }
}

// ---------------------------------------------------------------------------
extern "C" void kernel_launch(void* const* d_in, const int* in_sizes, int n_in,
                              void* d_out, int out_size, void* d_ws, size_t ws_size,
                              hipStream_t stream) {
    const float* x  = (const float*)d_in[0];
    const float* W1 = (const float*)d_in[1];
    const float* b1 = (const float*)d_in[2];
    const float* W2 = (const float*)d_in[3];
    const float* b2 = (const float*)d_in[4];
    const float* du = (const float*)d_in[5];
    const int*   ei = (const int*)d_in[6];
    int E = in_sizes[6] / 2;
    float* out = (float*)d_out;

    char* ws = (char*)d_ws;
    const size_t szAggF = (size_t)N_NODES * 256 * sizeof(float);   // 51.2 MB
    const size_t szAggB = (size_t)N_NODES * 256 * sizeof(bf16);    // 25.6 MB
    float* aggf = (float*)ws;               ws += szAggF;
    bf16*  aggb = (bf16*)ws;                ws += szAggB;
    bf16*  hbuf = (bf16*)ws;                ws += szAggB;
    bf16*  W1t  = (bf16*)ws;                ws += (size_t)256 * 256 * sizeof(bf16);
    bf16*  W2t  = (bf16*)ws;                ws += (size_t)NCLS_PAD * 256 * sizeof(bf16);

    const int n4 = N_NODES * 256 / 4;       // 3.2M float4 chunks
    const int cvtBlocks = (n4 + 255) / 256;
    const int edgeBlocks = (E + 3) / 4;

    // Weight prepack (cheap, rerun every call for determinism).
    prep_w1_kernel<<<256, 256, 0, stream>>>(W1, W1t);
    prep_w2_kernel<<<NCLS_PAD, 256, 0, stream>>>(W2, W2t);

    // ---- Layer 1 ----
    hipMemsetAsync(aggf, 0, szAggF, stream);
    scatter_f32_kernel<<<edgeBlocks, 256, 0, stream>>>(x, ei, E, aggf);
    cvt_f32_bf16_kernel<<<cvtBlocks, 256, 0, stream>>>(aggf, aggb, n4);
    gemm1_kernel<<<dim3(N_NODES / 16, 2), 256, 0, stream>>>(aggb, W1t, b1, du, hbuf);

    // ---- Layer 2 ----
    hipMemsetAsync(aggf, 0, szAggF, stream);
    scatter_bf16_kernel<<<edgeBlocks, 256, 0, stream>>>(hbuf, ei, E, aggf);
    cvt_f32_bf16_kernel<<<cvtBlocks, 256, 0, stream>>>(aggf, aggb, n4);
    gemm2_kernel<<<N_NODES / 16, 128, 0, stream>>>(aggb, W2t, b2, out);
}
